// InternLM3Attention_13340168422146
// MI455X (gfx1250) — matrix operand order repaired and checked by reading, not compiled, not run on hardware
//
#include <hip/hip_runtime.h>
#include <hip/hip_bf16.h>
#include <stdint.h>

typedef __bf16 v16bf __attribute__((ext_vector_type(16)));
typedef __bf16 v2bf  __attribute__((ext_vector_type(2)));
typedef float  v8f   __attribute__((ext_vector_type(8)));
typedef int    v4i   __attribute__((ext_vector_type(4)));

#define HID 4096
#define SEQ 2048
#define NH  32
#define NKV 8
#define HD  128

// native fptrunc f32->bf16 (RNE); backend picks the HW cvt instruction
__device__ __forceinline__ unsigned short f2bf(float f) {
  union { __bf16 b; unsigned short u; } c;
  c.b = (__bf16)f;
  return c.u;
}
__device__ __forceinline__ float bf2f(unsigned short h) {
  return __uint_as_float(((unsigned int)h) << 16);
}
// two f32 -> two bf16 packed in one dword
__device__ __forceinline__ unsigned int pack2bf(float x, float y) {
#if __has_builtin(__builtin_amdgcn_cvt_pk_bf16_f32)
  union { v2bf v; unsigned int u; } c;
  c.v = __builtin_amdgcn_cvt_pk_bf16_f32(x, y);
  return c.u;
#else
  union { v2bf v; unsigned int u; } c;
  c.v[0] = (__bf16)x;
  c.v[1] = (__bf16)y;
  return c.u;
#endif
}

#if defined(__gfx1250__) && __has_builtin(__builtin_amdgcn_global_load_async_to_lds_b128)
#define HAVE_ASYNC 1
typedef __attribute__((address_space(3))) v4i v4i_lds;
__device__ __forceinline__ void async_copy16(const void* g, void* l) {
  __builtin_amdgcn_global_load_async_to_lds_b128((v4i*)g, (v4i_lds*)l, 0, 0);
}
__device__ __forceinline__ void async_wait0() {
#if __has_builtin(__builtin_amdgcn_s_wait_asynccnt)
  __builtin_amdgcn_s_wait_asynccnt(0);
#else
  asm volatile("s_wait_asynccnt 0" ::: "memory");
#endif
}
#else
#define HAVE_ASYNC 0
__device__ __forceinline__ void async_copy16(const void* g, void* l) {
  *(uint4*)l = *(const uint4*)g;
}
__device__ __forceinline__ void async_wait0() {}
#endif

__device__ __forceinline__ v8f wmma_bf16(v16bf a, v16bf b, v8f c) {
  return __builtin_amdgcn_wmma_f32_16x16x32_bf16(false, a, false, b, (short)0, c,
                                                 false, false);
}

// A-matrix fragment (16x32 bf16): per half-lane the 16 K values are two
// contiguous 16B runs -> 2 x ds_load_b128.
__device__ __forceinline__ v16bf load_a_frag(const unsigned short* a, int lda, int lane) {
  union { v16bf v; uint4 q[2]; } r;
  const unsigned short* row = a + (lane & 15) * lda + ((lane >> 4) << 3);
  r.q[0] = *(const uint4*)(row);
  r.q[1] = *(const uint4*)(row + 16);
  return r.v;
}

// B-matrix fragment (32x16 bf16): lane = column (lane&15), K rows 8h..8h+7 and 16+8h..16+8h+7.
__device__ __forceinline__ v16bf load_b_frag(const unsigned short* b, int ldb, int lane) {
  union { v16bf v; unsigned short s[16]; } r;
  const unsigned short* col = b + (lane & 15) + ((lane >> 4) << 3) * ldb;
#pragma unroll
  for (int i = 0; i < 8; ++i) {
    r.s[i] = col[i * ldb];
    r.s[i + 8] = col[(i + 16) * ldb];
  }
  return r.v;
}

// ---------------------------------------------------------------- QKV GEMM
__global__ __launch_bounds__(256) void qkv_gemm_kernel(
    const float* __restrict__ hs, const float* __restrict__ wq,
    const float* __restrict__ wk, const float* __restrict__ wv,
    unsigned short* __restrict__ qws, unsigned short* __restrict__ kws,
    unsigned short* __restrict__ vws) {
  __shared__ unsigned short Alds[2][64][40];
  __shared__ unsigned short Blds[2][32][136];
  const int tid = threadIdx.x, lane = tid & 31, wave = tid >> 5;
  const int tm = blockIdx.x & 63, tn = blockIdx.x >> 6;
  const int m0 = tm << 6, n0 = tn << 7;

  const float* wbase;
  unsigned short* dst;
  int ldd, nrel;
  if (n0 < 4096)      { wbase = wq; dst = qws; ldd = 4096; nrel = n0; }
  else if (n0 < 5120) { wbase = wk; dst = kws; ldd = 1024; nrel = n0 - 4096; }
  else                { wbase = wv; dst = vws; ldd = 1024; nrel = n0 - 5120; }

  const int am = tid >> 2, ak = (tid & 3) << 3;
  const float* asrc = hs + (size_t)(m0 + am) * HID + ak;

  const int bn2 = (tid & 63) << 1, bkk = (tid >> 6) << 3;
  const float* wrow0 = wbase + (size_t)(nrel + bn2) * HID + bkk;
  const float* wrow1 = wrow0 + HID;

  const int wm = (wave >> 2) << 5, wn = (wave & 3) << 5;

  auto stage = [&](int k0, int buf) {
    float4 f0 = *(const float4*)(asrc + k0);
    float4 f1 = *(const float4*)(asrc + k0 + 4);
    uint4 st;
    st.x = pack2bf(f0.x, f0.y); st.y = pack2bf(f0.z, f0.w);
    st.z = pack2bf(f1.x, f1.y); st.w = pack2bf(f1.z, f1.w);
    *(uint4*)&Alds[buf][am][ak] = st;
    float4 a0 = *(const float4*)(wrow0 + k0);
    float4 a1 = *(const float4*)(wrow0 + k0 + 4);
    float4 b0 = *(const float4*)(wrow1 + k0);
    float4 b1 = *(const float4*)(wrow1 + k0 + 4);
    *(unsigned int*)&Blds[buf][bkk + 0][bn2] = pack2bf(a0.x, b0.x);
    *(unsigned int*)&Blds[buf][bkk + 1][bn2] = pack2bf(a0.y, b0.y);
    *(unsigned int*)&Blds[buf][bkk + 2][bn2] = pack2bf(a0.z, b0.z);
    *(unsigned int*)&Blds[buf][bkk + 3][bn2] = pack2bf(a0.w, b0.w);
    *(unsigned int*)&Blds[buf][bkk + 4][bn2] = pack2bf(a1.x, b1.x);
    *(unsigned int*)&Blds[buf][bkk + 5][bn2] = pack2bf(a1.y, b1.y);
    *(unsigned int*)&Blds[buf][bkk + 6][bn2] = pack2bf(a1.z, b1.z);
    *(unsigned int*)&Blds[buf][bkk + 7][bn2] = pack2bf(a1.w, b1.w);
  };

  v8f acc[2][2] = {};
  stage(0, 0);
  const int nit = HID / 32;
  for (int it = 0; it < nit; ++it) {
    const int cur = it & 1;
    __syncthreads();
    if (it + 1 < nit) stage((it + 1) << 5, cur ^ 1);
    v16bf a0 = load_a_frag(&Alds[cur][wm][0], 40, lane);
    v16bf a1 = load_a_frag(&Alds[cur][wm + 16][0], 40, lane);
    v16bf b0 = load_b_frag(&Blds[cur][0][wn], 136, lane);
    v16bf b1 = load_b_frag(&Blds[cur][0][wn + 16], 136, lane);
    acc[0][0] = wmma_bf16(a0, b0, acc[0][0]);
    acc[0][1] = wmma_bf16(a0, b1, acc[0][1]);
    acc[1][0] = wmma_bf16(a1, b0, acc[1][0]);
    acc[1][1] = wmma_bf16(a1, b1, acc[1][1]);
  }
  const int rbase = (lane >> 4) << 3, cb = lane & 15;
#pragma unroll
  for (int i = 0; i < 2; ++i)
#pragma unroll
    for (int j = 0; j < 2; ++j) {
      unsigned short* p =
          dst + (size_t)(m0 + wm + (i << 4) + rbase) * ldd + nrel + wn + (j << 4) + cb;
#pragma unroll
      for (int r = 0; r < 8; r += 2) {  // one packed cvt feeds two b16 stores
        unsigned int pp = pack2bf(acc[i][j][r], acc[i][j][r + 1]);
        p[(size_t)r * ldd]       = (unsigned short)pp;
        p[(size_t)(r + 1) * ldd] = (unsigned short)(pp >> 16);
      }
    }
}

// ---------------------------------------------------------------- RoPE
__global__ void rope_kernel(unsigned short* __restrict__ qws,
                            unsigned short* __restrict__ kws) {
  const int NQ = 4096 * NH * 64;
  const int NT = NQ + 4096 * NKV * 64;
  for (int idx = blockIdx.x * blockDim.x + threadIdx.x; idx < NT;
       idx += gridDim.x * blockDim.x) {
    unsigned short* base;
    int row, j;
    if (idx < NQ) {
      row = idx >> 11; int rem = idx & 2047;
      base = qws + (size_t)row * 4096 + ((rem >> 6) << 7);
      j = rem & 63;
    } else {
      int t = idx - NQ;
      row = t >> 9; int rem = t & 511;
      base = kws + (size_t)row * 1024 + ((rem >> 6) << 7);
      j = rem & 63;
    }
    int s = row & (SEQ - 1);
    float ang = (float)s * __expf(-(float)j * 0.14391157f);  // 10000^(-j/64)
    float sn, cs;
    __sincosf(ang, &sn, &cs);
    float x1 = bf2f(base[j]), x2 = bf2f(base[j + 64]);
    unsigned int pp = pack2bf(x1 * cs - x2 * sn, x2 * cs + x1 * sn);
    base[j]      = (unsigned short)pp;
    base[j + 64] = (unsigned short)(pp >> 16);
  }
}

// ---------------------------------------------------------------- Flash attention
__global__ __launch_bounds__(256) void attn_kernel(
    const unsigned short* __restrict__ qws, const unsigned short* __restrict__ kws,
    const unsigned short* __restrict__ vws, unsigned short* __restrict__ aws) {
  __shared__ unsigned short KT[2][128][40];
  __shared__ unsigned short Vl[2][32][136];
  __shared__ unsigned short Pl[4][16][40];
  const int tid = threadIdx.x, lane = tid & 31, wave = tid >> 5;
  const int qblk = blockIdx.x & 31;
  const int h    = (blockIdx.x >> 5) & 31;
  const int b    = blockIdx.x >> 10;
  const int hkv  = h >> 2;
  const int q0   = qblk << 6;
  const int qw   = q0 + (wave << 4);

  v16bf qfrag[4];
  {
    const unsigned short* qrow =
        qws + ((size_t)b * SEQ + qw + (lane & 15)) * 4096 + h * HD + ((lane >> 4) << 3);
#pragma unroll
    for (int c = 0; c < 4; ++c) {
      union { v16bf v; uint4 q[2]; } r;
      r.q[0] = *(const uint4*)(qrow + (c << 5));
      r.q[1] = *(const uint4*)(qrow + (c << 5) + 16);
      qfrag[c] = r.v;
    }
  }

  v8f o[8] = {};
  float mrun[8], lrun[8], alph[8];
#pragma unroll
  for (int r = 0; r < 8; ++r) { mrun[r] = -3.0e38f; lrun[r] = 0.0f; }

  const int k2 = (tid & 15) << 1, kds = (tid >> 4) << 3;
  const int vkk = tid >> 3, vds = (tid & 7) << 4;
  const float scale = 0.08838834764831845f;  // 1/sqrt(128)
  const int wave_qmax = qw + 15;
  const int nit = (q0 + 64) >> 5;

  auto stage = [&](int kb, int buf) {
    size_t kbase = ((size_t)b * SEQ + kb + k2) * 1024 + hkv * HD + kds;
    union { uint4 q; unsigned short s[8]; } r0, r1;
    r0.q = *(const uint4*)(kws + kbase);
    r1.q = *(const uint4*)(kws + kbase + 1024);
#pragma unroll
    for (int i = 0; i < 8; ++i)
      *(unsigned int*)&KT[buf][kds + i][k2] =
          (unsigned int)r0.s[i] | ((unsigned int)r1.s[i] << 16);
    size_t vbase = ((size_t)b * SEQ + kb + vkk) * 1024 + hkv * HD + vds;
    async_copy16(vws + vbase, &Vl[buf][vkk][vds]);
    async_copy16(vws + vbase + 8, &Vl[buf][vkk][vds + 8]);
  };

  stage(0, 0);
  for (int it = 0; it < nit; ++it) {
    const int cur = it & 1, kb = it << 5;
    async_wait0();
    __syncthreads();
    if (it + 1 < nit) stage(kb + 32, cur ^ 1);
    if (kb <= wave_qmax) {  // causal block cull (kb <= qw guaranteed)
      v8f s0 = {}, s1 = {};
#pragma unroll
      for (int c = 0; c < 4; ++c) {
        v16bf bk0 = load_b_frag(&KT[cur][c << 5][0], 40, lane);
        v16bf bk1 = load_b_frag(&KT[cur][c << 5][16], 40, lane);
        s0 = wmma_bf16(qfrag[c], bk0, s0);
        s1 = wmma_bf16(qfrag[c], bk1, s1);
      }
      const int colb = kb + (lane & 15);
      const int rowb = qw + ((lane >> 4) << 3);
      const int prow = (lane >> 4) << 3;
      const int pcol = lane & 15;
#pragma unroll
      for (int r = 0; r < 8; ++r) {
        int rp = rowb + r;
        float a0 = (colb <= rp)      ? s0[r] * scale : -3.0e38f;
        float a1 = (colb + 16 <= rp) ? s1[r] * scale : -3.0e38f;
        float mx = fmaxf(a0, a1);
#pragma unroll
        for (int off = 1; off < 16; off <<= 1)
          mx = fmaxf(mx, __shfl_xor(mx, off, 32));
        float mnew = fmaxf(mrun[r], mx);
        float al = __expf(mrun[r] - mnew);
        float p0 = __expf(a0 - mnew);
        float p1 = __expf(a1 - mnew);
        float rs = p0 + p1;
#pragma unroll
        for (int off = 1; off < 16; off <<= 1)
          rs += __shfl_xor(rs, off, 32);
        lrun[r] = lrun[r] * al + rs;
        mrun[r] = mnew;
        alph[r] = al;
        unsigned int pp = pack2bf(p0, p1);  // one packed cvt, two b16 stores
        Pl[wave][prow + r][pcol]      = (unsigned short)pp;
        Pl[wave][prow + r][pcol + 16] = (unsigned short)(pp >> 16);
      }
#pragma unroll
      for (int t = 0; t < 8; ++t)
#pragma unroll
        for (int r = 0; r < 8; ++r) o[t][r] *= alph[r];
      asm volatile("s_wait_dscnt 0" ::: "memory");  // same-wave LDS in-order
      v16bf pf = load_a_frag(&Pl[wave][0][0], 40, lane);
#pragma unroll
      for (int t = 0; t < 8; ++t) {
        v16bf vf = load_b_frag(&Vl[cur][0][t << 4], 136, lane);
        o[t] = wmma_bf16(pf, vf, o[t]);
      }
    }
  }
  const int prow = (lane >> 4) << 3, pcol = lane & 15;
#pragma unroll
  for (int r = 0; r < 8; ++r) lrun[r] = 1.0f / lrun[r];
#pragma unroll
  for (int t = 0; t < 8; ++t) {
    unsigned short* p =
        aws + ((size_t)b * SEQ + qw + prow) * 4096 + h * HD + (t << 4) + pcol;
#pragma unroll
    for (int r = 0; r < 8; r += 2) {
      unsigned int pp = pack2bf(o[t][r] * lrun[r], o[t][r + 1] * lrun[r + 1]);
      p[(size_t)r * 4096]       = (unsigned short)pp;
      p[(size_t)(r + 1) * 4096] = (unsigned short)(pp >> 16);
    }
  }
}

// ---------------------------------------------------------------- Output GEMM
__global__ __launch_bounds__(256) void out_gemm_kernel(
    const unsigned short* __restrict__ attn, const float* __restrict__ wo,
    float* __restrict__ out) {
  __shared__ unsigned short Alds[2][64][40];
  __shared__ unsigned short Blds[2][32][136];
  const int tid = threadIdx.x, lane = tid & 31, wave = tid >> 5;
  const int tm = blockIdx.x & 63, tn = blockIdx.x >> 6;
  const int m0 = tm << 6, n0 = tn << 7;

  const int am = tid >> 2, ak = (tid & 3) << 3;
  const unsigned short* asrc = attn + (size_t)(m0 + am) * HID + ak;

  const int bn2 = (tid & 63) << 1, bkk = (tid >> 6) << 3;
  const float* wrow0 = wo + (size_t)(n0 + bn2) * HID + bkk;
  const float* wrow1 = wrow0 + HID;

  const int wm = (wave >> 2) << 5, wn = (wave & 3) << 5;

  auto stage = [&](int k0, int buf) {
    async_copy16(asrc + k0, &Alds[buf][am][ak]);  // bf16 tile -> async path
    float4 a0 = *(const float4*)(wrow0 + k0);
    float4 a1 = *(const float4*)(wrow0 + k0 + 4);
    float4 b0 = *(const float4*)(wrow1 + k0);
    float4 b1 = *(const float4*)(wrow1 + k0 + 4);
    *(unsigned int*)&Blds[buf][bkk + 0][bn2] = pack2bf(a0.x, b0.x);
    *(unsigned int*)&Blds[buf][bkk + 1][bn2] = pack2bf(a0.y, b0.y);
    *(unsigned int*)&Blds[buf][bkk + 2][bn2] = pack2bf(a0.z, b0.z);
    *(unsigned int*)&Blds[buf][bkk + 3][bn2] = pack2bf(a0.w, b0.w);
    *(unsigned int*)&Blds[buf][bkk + 4][bn2] = pack2bf(a1.x, b1.x);
    *(unsigned int*)&Blds[buf][bkk + 5][bn2] = pack2bf(a1.y, b1.y);
    *(unsigned int*)&Blds[buf][bkk + 6][bn2] = pack2bf(a1.z, b1.z);
    *(unsigned int*)&Blds[buf][bkk + 7][bn2] = pack2bf(a1.w, b1.w);
  };

  v8f acc[2][2] = {};
  stage(0, 0);
  const int nit = HID / 32;
  for (int it = 0; it < nit; ++it) {
    const int cur = it & 1;
    async_wait0();
    __syncthreads();
    if (it + 1 < nit) stage((it + 1) << 5, cur ^ 1);
    v16bf a0 = load_a_frag(&Alds[cur][wm][0], 40, lane);
    v16bf a1 = load_a_frag(&Alds[cur][wm + 16][0], 40, lane);
    v16bf b0 = load_b_frag(&Blds[cur][0][wn], 136, lane);
    v16bf b1 = load_b_frag(&Blds[cur][0][wn + 16], 136, lane);
    acc[0][0] = wmma_bf16(a0, b0, acc[0][0]);
    acc[0][1] = wmma_bf16(a0, b1, acc[0][1]);
    acc[1][0] = wmma_bf16(a1, b0, acc[1][0]);
    acc[1][1] = wmma_bf16(a1, b1, acc[1][1]);
  }
  const int rbase = (lane >> 4) << 3, cb = lane & 15;
#pragma unroll
  for (int i = 0; i < 2; ++i)
#pragma unroll
    for (int j = 0; j < 2; ++j) {
      float* p = out + (size_t)(m0 + wm + (i << 4) + rbase) * HID + n0 + wn + (j << 4) + cb;
#pragma unroll
      for (int r = 0; r < 8; ++r) p[(size_t)r * HID] = acc[i][j][r];
    }
}

extern "C" void kernel_launch(void* const* d_in, const int* in_sizes, int n_in,
                              void* d_out, int out_size, void* d_ws, size_t ws_size,
                              hipStream_t stream) {
  (void)in_sizes; (void)n_in; (void)out_size; (void)ws_size;
  const float* hs = (const float*)d_in[0];
  const float* wq = (const float*)d_in[1];
  const float* wk = (const float*)d_in[2];
  const float* wv = (const float*)d_in[3];
  const float* wo = (const float*)d_in[4];
  float* out = (float*)d_out;

  char* ws = (char*)d_ws;
  unsigned short* qws = (unsigned short*)(ws);             // 4096*4096 bf16
  unsigned short* kws = (unsigned short*)(ws + 33554432);  // 4096*1024 bf16
  unsigned short* vws = (unsigned short*)(ws + 41943040);  // 4096*1024 bf16
  unsigned short* aws = (unsigned short*)(ws + 50331648);  // 4096*4096 bf16

  qkv_gemm_kernel<<<dim3(64 * 48), dim3(256), 0, stream>>>(hs, wq, wk, wv,
                                                           qws, kws, vws);
  rope_kernel<<<dim3(4096), dim3(256), 0, stream>>>(qws, kws);
  attn_kernel<<<dim3(2 * 32 * 32), dim3(256), 0, stream>>>(qws, kws, vws, aws);
  out_gemm_kernel<<<dim3(64 * 32), dim3(256), 0, stream>>>(aws, wo, out);
}
